// VTIRTOld_84791244357666
// MI455X (gfx1250) — matrix operand order repaired and checked by reading, not compiled, not run on hardware
//
#include <hip/hip_runtime.h>
#include <hip/hip_bf16.h>

typedef __attribute__((ext_vector_type(16))) _Float16 v16h;
typedef __attribute__((ext_vector_type(8)))  _Float16 v8h;
typedef __attribute__((ext_vector_type(8)))  float    v8f;

#define UU 4096
#define TT 200
#define KK 64
#define QQ 10000
#define HH 128

// Branchless exact-erf GELU: Abramowitz-Stegun 7.1.26 rational erf
// (max abs err ~1.5e-7) on hardware TRANS ops (v_rcp_f32, v_exp_f32).
// No exec-mask branches -> no s_and_saveexec churn, co-executes with WMMA.
__device__ __forceinline__ float gelu_exact(float x) {
    const float y  = x * 0.70710678118654752f;     // x / sqrt(2)
    const float ay = fabsf(y);
    const float t  = __builtin_amdgcn_rcpf(1.0f + 0.3275911f * ay);
    float poly = 1.061405429f;
    poly = poly * t - 1.453152027f;
    poly = poly * t + 1.421413741f;
    poly = poly * t - 0.284496736f;
    poly = poly * t + 0.254829592f;
    poly = poly * t;
    const float e  = __builtin_amdgcn_exp2f(-ay * ay * 1.4426950408889634f);
    const float er = copysignf(1.0f - poly * e, y);
    return 0.5f * x * (1.0f + er);
}

// ---------------------------------------------------------------------------
// Kernel 0: precompute kc_of_q[q] = argmax_k kmap[q][k]  (one-hot -> first set)
// ---------------------------------------------------------------------------
__global__ __launch_bounds__(256) void vtirt_kc_kernel(
    const unsigned char* __restrict__ kmap, int* __restrict__ kc_of_q) {
    int q = blockIdx.x * 256 + threadIdx.x;
    if (q >= QQ) return;
    int kc = 0;
    for (int k = 0; k < KK; ++k) {
        if (kmap[q * KK + k]) { kc = k; break; }
    }
    kc_of_q[q] = kc;
}

// ---------------------------------------------------------------------------
// Kernel 1: MLP over all U*T trials.  One wave = 16 trials (one WMMA M-tile).
//   layer1 (3->128)  : VALU, each lane computes exactly the K-halves its
//                      A-fragment needs (no LDS staging for h1)
//   layer2 (128->128): v_wmma_f32_16x16x32_f16, W2 (f16, transposed) in LDS
//   layer3 (128->2)  : folded into the N-tile loop, shfl_xor reduction
// Writes r[t*U+u], mu[t*U+u] to workspace (T,U layout for the scans).
// ---------------------------------------------------------------------------
#define W2PITCH 136   // halfs per w2t row (pad to dodge LDS bank conflicts)

__global__ __launch_bounds__(256) void vtirt_mlp_kernel(
    const int* __restrict__ q_id, const int* __restrict__ resp,
    const float* __restrict__ diff_mu, const float* __restrict__ disc_mu,
    const float* __restrict__ W1, const float* __restrict__ b1,
    const float* __restrict__ W2, const float* __restrict__ b2,
    const float* __restrict__ W3, const float* __restrict__ b3,
    float* __restrict__ r_ws, float* __restrict__ mu_ws) {

    __shared__ _Float16 w2t[HH * W2PITCH];   // [n][k] transposed, f16  (~34.8 KB)
    __shared__ float    w1s[3 * HH];
    __shared__ float    b1s[HH];
    __shared__ float    b2s[HH];
    __shared__ float    w3s[HH * 2];
    __shared__ float    b3s[2];

    const int tid = threadIdx.x;

    // Cooperative load: W2 -> transposed f16 in LDS (coalesced global reads).
    for (int idx = tid; idx < HH * HH; idx += 256) {
        int n = idx & 127, k = idx >> 7;
        w2t[n * W2PITCH + k] = (_Float16)W2[k * HH + n];
    }
    for (int idx = tid; idx < 3 * HH; idx += 256) w1s[idx] = W1[idx];
    for (int idx = tid; idx < HH; idx += 256) { b1s[idx] = b1[idx]; b2s[idx] = b2[idx]; }
    for (int idx = tid; idx < 2 * HH; idx += 256) w3s[idx] = W3[idx];
    if (tid < 2) b3s[tid] = b3[tid];
    __syncthreads();

    const int wave    = tid >> 5;
    const int lane    = tid & 31;
    const int m       = lane & 15;        // row M of this lane's A fragment
    const int hiHalf  = lane >> 4;        // 0: low K half, 1: high K half
    const int koff8   = hiHalf * 8;

    const int s = blockIdx.x * 128 + wave * 16 + m;   // trial index (U*T = 6400*128)
    const int u = s / TT, t = s % TT;
    const int q = q_id[u * TT + t];
    const float td = diff_mu[q];
    const float ts = disc_mu[q];
    const float rp = (float)resp[u * TT + t];

    // ---- layer 1: build A fragments (16x32 f16, 4 K-steps) in registers ----
    // ISA 16-bit A layout: lane L holds row M=L&15; half i -> K(i) = i + (i>=8?8:0) + (L>=16?8:0)
    v16h afrag[4];
#pragma unroll
    for (int g = 0; g < 4; ++g) {
#pragma unroll
        for (int i = 0; i < 16; ++i) {
            int kk = g * 32 + i + ((i >= 8) ? 8 : 0) + koff8;
            float h = td * w1s[kk] + ts * w1s[HH + kk] + rp * w1s[2 * HH + kk] + b1s[kk];
            afrag[g][i] = (_Float16)gelu_exact(h);
        }
    }

    // ---- layer 2 (WMMA) + fused layer-3 partial dot products ----
    const int colLane = lane & 15;        // column N within tile (C/B layout)
    const int koffB   = hiHalf * 8;       // B layout: lanes 16-31 hold K+8..K+15, K+24..K+31
    float p0[8], p1[8];
#pragma unroll
    for (int r = 0; r < 8; ++r) { p0[r] = 0.0f; p1[r] = 0.0f; }

#pragma unroll
    for (int nt = 0; nt < 8; ++nt) {
        const int col = nt * 16 + colLane;
        const float bias = b2s[col];
        v8f c;
#pragma unroll
        for (int r = 0; r < 8; ++r) c[r] = bias;

#pragma unroll
        for (int g = 0; g < 4; ++g) {
            const _Float16* bp = &w2t[col * W2PITCH + g * 32 + koffB];
            union { v16h v; v8h h[2]; } bu;
            bu.h[0] = *(const v8h*)bp;
            bu.h[1] = *(const v8h*)(bp + 16);
            c = __builtin_amdgcn_wmma_f32_16x16x32_f16(
                    false, afrag[g], false, bu.v, (short)0, c, false, false);
        }
        // gelu + layer-3 partials: lane holds rows {0..7}+8*hiHalf, column col
        const float w3a = w3s[2 * col], w3b = w3s[2 * col + 1];
#pragma unroll
        for (int r = 0; r < 8; ++r) {
            float h2 = gelu_exact(c[r]);
            p0[r] += h2 * w3a;
            p1[r] += h2 * w3b;
        }
    }

    // Butterfly-reduce layer-3 partials across the 16 lanes of each K/row half.
#pragma unroll
    for (int msk = 1; msk < 16; msk <<= 1) {
#pragma unroll
        for (int r = 0; r < 8; ++r) {
            p0[r] += __shfl_xor(p0[r], msk, 32);
            p1[r] += __shfl_xor(p1[r], msk, 32);
        }
    }

    // Writeback: lanes 0-7 own rows 0-7, lanes 16-23 own rows 8-15.
    if ((lane & 15) < 8) {
        const int r = lane & 7;
        const int mm = r + hiHalf * 8;
        float a0 = b3s[0], a1 = b3s[1];
#pragma unroll
        for (int rr = 0; rr < 8; ++rr) {           // cndmask select p[r] (no scratch)
            bool sel = (rr == r);
            a0 = sel ? (p0[rr] + b3s[0]) : a0;
            a1 = sel ? (p1[rr] + b3s[1]) : a1;
        }
        const float muv = gelu_exact(a0);
        const float lv  = gelu_exact(a1);
        // r = (1 / max(exp(0.5*lv), 1e-8))^2  ==  min(exp(-lv), 1e16)
        const float rr2 = fminf(__builtin_amdgcn_exp2f(-lv * 1.4426950408889634f), 1e16f);

        const int s2 = blockIdx.x * 128 + wave * 16 + mm;
        const int u2 = s2 / TT, t2 = s2 % TT;
        r_ws[t2 * UU + u2]  = rr2;
        mu_ws[t2 * UU + u2] = muv;
    }
}

// ---------------------------------------------------------------------------
// Kernel 2: backward Kalman scan (in-place r->b, mu->c) + forward KC scan.
// One user per thread; per-user 64-float KC state lives in LDS (pitch 65).
// ---------------------------------------------------------------------------
__global__ __launch_bounds__(64) void vtirt_scan_kernel(
    const int* __restrict__ q_id, const int* __restrict__ kc_of_q,
    const float* __restrict__ diff_mu, const float* __restrict__ disc_mu,
    float* __restrict__ b_ws, float* __restrict__ c_ws,
    float* __restrict__ out) {

    __shared__ float st[64 * 65];
    const int tid = threadIdx.x;
    const int u   = blockIdx.x * 64 + tid;

    // backward: b_t = 1/(1 + r_t + (1 - b_{t+1})), c_t = b_t*(c_{t+1} + r_t*mu_t)
    float bn = 1.0f, cn = 0.0f;
    for (int t = TT - 1; t >= 0; --t) {
        const float rt = b_ws[t * UU + u];   // still holds r
        const float mt = c_ws[t * UU + u];   // still holds mu
        const float bt = 1.0f / (2.0f + rt - bn);
        const float ct = bt * (cn + rt * mt);
        b_ws[t * UU + u] = bt;
        c_ws[t * UU + u] = ct;
        bn = bt; cn = ct;
    }

    // forward: gather active-KC ability, update, scatter
#pragma unroll 4
    for (int k = 0; k < KK; ++k) st[tid * 65 + k] = 0.0f;
    for (int t = 0; t < TT; ++t) {
        const int q  = q_id[u * TT + t];
        const int kc = kc_of_q[q];
        const float bt = b_ws[t * UU + u];
        const float ct = c_ws[t * UU + u];
        const float prev = st[tid * 65 + kc];
        const float mt = bt * prev + ct;
        st[tid * 65 + kc] = mt;
        out[u * TT + t] = disc_mu[q] * (mt - diff_mu[q]);   // trial_logits
    }
#pragma unroll 4
    for (int k = 0; k < KK; ++k)
        out[UU * TT + u * KK + k] = st[tid * 65 + k];       // last_ability_kc
}

// ---------------------------------------------------------------------------
extern "C" void kernel_launch(void* const* d_in, const int* in_sizes, int n_in,
                              void* d_out, int out_size, void* d_ws, size_t ws_size,
                              hipStream_t stream) {
    (void)in_sizes; (void)n_in; (void)out_size; (void)ws_size;
    // setup_inputs order: mask, q_id, kmap, resp, diff_mu, disc_mu, W1,b1,W2,b2,W3,b3
    const int*           q_id    = (const int*)d_in[1];
    const unsigned char* kmap    = (const unsigned char*)d_in[2];
    const int*           resp    = (const int*)d_in[3];
    const float*         diff_mu = (const float*)d_in[4];
    const float*         disc_mu = (const float*)d_in[5];
    const float*         W1      = (const float*)d_in[6];
    const float*         b1      = (const float*)d_in[7];
    const float*         W2      = (const float*)d_in[8];
    const float*         b2      = (const float*)d_in[9];
    const float*         W3      = (const float*)d_in[10];
    const float*         b3      = (const float*)d_in[11];

    float* r_ws    = (float*)d_ws;                    // (T,U) -> becomes b_seq
    float* mu_ws   = r_ws + (size_t)UU * TT;          // (T,U) -> becomes c_seq
    int*   kc_of_q = (int*)(mu_ws + (size_t)UU * TT); // (Q,)
    float* out     = (float*)d_out;                   // [U*T logits | U*K abilities]

    vtirt_kc_kernel<<<(QQ + 255) / 256, 256, 0, stream>>>(kmap, kc_of_q);
    vtirt_mlp_kernel<<<(UU * TT) / 128, 256, 0, stream>>>(
        q_id, resp, diff_mu, disc_mu, W1, b1, W2, b2, W3, b3, r_ws, mu_ws);
    vtirt_scan_kernel<<<UU / 64, 64, 0, stream>>>(
        q_id, kc_of_q, diff_mu, disc_mu, r_ws, mu_ws, out);
}
